// DynGCN_26336739459145
// MI455X (gfx1250) — matrix operand order repaired and checked by reading, not compiled, not run on hardware
//
#include <hip/hip_runtime.h>
#include <hip/hip_bf16.h>

#define HDIM 128
#define TSTEPS 12

typedef __attribute__((ext_vector_type(16))) __bf16 v16bf;
typedef __attribute__((ext_vector_type(8)))  float  v8f;

#if defined(__has_builtin)
#if __has_builtin(__builtin_amdgcn_global_load_async_to_lds_b128) && \
    __has_builtin(__builtin_amdgcn_s_wait_asynccnt)
#define USE_ASYNC_LDS 1
#endif
#endif

#ifdef USE_ASYNC_LDS
typedef __attribute__((ext_vector_type(4))) int i4v;
typedef __attribute__((address_space(1))) i4v gi4;   // global int4*
typedef __attribute__((address_space(3))) i4v li4;   // LDS int4*
#endif

__device__ __forceinline__ unsigned short f32_to_bf16(float f) {
    unsigned u = __float_as_uint(f);
    unsigned r = u + 0x7FFFu + ((u >> 16) & 1u);   // round-to-nearest-even
    return (unsigned short)(r >> 16);
}

// ---------------------------------------------------------------------------
// Degree / normalization kernels
// ---------------------------------------------------------------------------
__global__ void init_deg_kernel(float* __restrict__ deg, int n) {
    int i = blockIdx.x * blockDim.x + threadIdx.x;
    if (i < n) deg[i] = 1.0f;                       // self-loop
}

__global__ void deg_edges_kernel(float* __restrict__ deg,
                                 const int* __restrict__ dst, int ne) {
    int e = blockIdx.x * blockDim.x + threadIdx.x;
    if (e < ne) atomicAdd(&deg[dst[e]], 1.0f);
}

__global__ void dinv_kernel(const float* __restrict__ deg,
                            float* __restrict__ dinv, int n) {
    int i = blockIdx.x * blockDim.x + threadIdx.x;
    if (i < n) dinv[i] = rsqrtf(deg[i]);
}

// ---------------------------------------------------------------------------
// WMMA GEMM: Y[nrows x 128] = act(X)[nrows x 128] @ W[128 x 128]
// Block = 256 threads (8 waves). Block tile = 128 rows x 128 cols.
// W staged into LDS in B-fragment layout. X tile staged via async DMA
// (global->LDS, fp32) when available; bf16 conversion done at fragment
// assembly. Fallback: sync stage to a padded bf16 tile.
// ---------------------------------------------------------------------------
__launch_bounds__(256)
__global__ void gcn_gemm_wmma(const float* __restrict__ X,
                              const float* __restrict__ W,
                              float* __restrict__ Y,
                              int nrows, int relu_in)
{
    __shared__ __align__(32) unsigned short ldsW[4 * 8 * 32 * 16]; // [kw][nt][lane][16]
#ifdef USE_ASYNC_LDS
    __shared__ __align__(16) float ldsXf[128 * 132];               // padded fp32 tile
#else
    __shared__ __align__(32) unsigned short ldsX[128 * 136];       // padded bf16 tile
#endif

    const int tid = threadIdx.x;
    const long rowbase = (long)blockIdx.x * 128;

#ifdef USE_ASYNC_LDS
    // ---- async DMA: global fp32 tile -> LDS, no VGPR round trip ----
    #pragma unroll
    for (int i = 0; i < 16; ++i) {
        int idx = tid + i * 256;                      // 4096 float4 slots
        int row = idx >> 5;
        int c4  = (idx & 31) << 2;
        long grow = rowbase + row;
        if (grow >= nrows) grow = nrows - 1;          // clamp; OOB rows never stored
        __builtin_amdgcn_global_load_async_to_lds_b128(
            (gi4*)(X + grow * HDIM + c4),
            (li4*)(&ldsXf[row * 132 + c4]), 0, 0);
    }
#endif

    // ---- stage W (128x128 f32) into exact B-fragment layout (bf16) ----
    for (int idx = tid; idx < 4 * 8 * 32 * 16; idx += 256) {
        int j    = idx & 15;
        int lane = (idx >> 4) & 31;
        int nt   = (idx >> 9) & 7;
        int kw   = (idx >> 12) & 3;
        int k = kw * 32 + ((lane & 16) ? 8 : 0) + (j & 7) + ((j & 8) ? 16 : 0);  // lanes 0-15: K 0-7,16-23; 16-31: K 8-15,24-31
        int n = nt * 16 + (lane & 15);
        ldsW[idx] = f32_to_bf16(W[k * HDIM + n]);
    }

#ifdef USE_ASYNC_LDS
    __builtin_amdgcn_s_wait_asynccnt(0);
#else
    // ---- sync stage 128-row X tile (coalesced float4), optional ReLU ----
    #pragma unroll
    for (int i = 0; i < 16; ++i) {
        int idx = tid + i * 256;
        int row = idx >> 5;
        int c4  = (idx & 31) << 2;
        long grow = rowbase + row;
        if (grow >= nrows) grow = nrows - 1;
        const float4 v = *(const float4*)(X + grow * HDIM + c4);
        float a0 = v.x, a1 = v.y, a2 = v.z, a3 = v.w;
        if (relu_in) {
            a0 = fmaxf(a0, 0.f); a1 = fmaxf(a1, 0.f);
            a2 = fmaxf(a2, 0.f); a3 = fmaxf(a3, 0.f);
        }
        unsigned short* p = &ldsX[row * 136 + c4];
        p[0] = f32_to_bf16(a0); p[1] = f32_to_bf16(a1);
        p[2] = f32_to_bf16(a2); p[3] = f32_to_bf16(a3);
    }
#endif
    __syncthreads();

    const int wave = tid >> 5;
    const int lane = tid & 31;
    const int lrow = wave * 16 + (lane & 15);         // block-local A row
    const int lh   = (lane >> 4) & 1;                 // lane half

    v8f acc[8];
    const v8f zero = {0.f, 0.f, 0.f, 0.f, 0.f, 0.f, 0.f, 0.f};
    #pragma unroll
    for (int nt = 0; nt < 8; ++nt) acc[nt] = zero;

    #pragma unroll
    for (int kw = 0; kw < 4; ++kw) {
        // A-fragment: per ISA 16-bit A 16x32 layout, each lane's 16 halves are
        // two contiguous 8-element runs: K = kbase..kbase+7 and K+16..K+23,
        // kbase = kw*32 + (laneHalf ? 8 : 0).
        union { unsigned short h[16]; uint4 u[2]; v16bf a; } af;
#ifdef USE_ASYNC_LDS
        {
            const float* xr = &ldsXf[lrow * 132 + kw * 32 + lh * 8];
            float4 f0 = *(const float4*)(xr);
            float4 f1 = *(const float4*)(xr + 4);
            float4 f2 = *(const float4*)(xr + 16);
            float4 f3 = *(const float4*)(xr + 20);
            float fv[16] = { f0.x, f0.y, f0.z, f0.w,  f1.x, f1.y, f1.z, f1.w,
                             f2.x, f2.y, f2.z, f2.w,  f3.x, f3.y, f3.z, f3.w };
            #pragma unroll
            for (int j = 0; j < 16; ++j) {
                float fj = fv[j];
                if (relu_in) fj = fmaxf(fj, 0.f);
                af.h[j] = f32_to_bf16(fj);
            }
        }
#else
        {
            const unsigned short* xr = &ldsX[lrow * 136 + kw * 32 + lh * 8];
            af.u[0] = *(const uint4*)(xr);
            af.u[1] = *(const uint4*)(xr + 16);
        }
#endif
        #pragma unroll
        for (int nt = 0; nt < 8; ++nt) {
            const v16bf b = *(const v16bf*)&ldsW[((((kw << 3) | nt) * 32) + lane) << 4];
            acc[nt] = __builtin_amdgcn_wmma_f32_16x16x32_bf16(
                false, af.a, false, b, (short)0, acc[nt], false, false);
        }
    }

    // ---- store D: VGPR v -> row v (+8 for lanes 16-31), col = lane%16 ----
    const long gr0  = rowbase + wave * 16 + (lh ? 8 : 0);
    const int  col0 = lane & 15;
    #pragma unroll
    for (int nt = 0; nt < 8; ++nt) {
        #pragma unroll
        for (int v = 0; v < 8; ++v) {
            long r = gr0 + v;
            if (r < nrows) Y[r * HDIM + nt * 16 + col0] = acc[nt][v];
        }
    }
}

// ---------------------------------------------------------------------------
// Scatter init: out[i][c] = h[i][c] * dinv[i]^2 (self loop) + bias[c]
// ---------------------------------------------------------------------------
__global__ void scatter_init_kernel(const float* __restrict__ h,
                                    const float* __restrict__ dinv,
                                    const float* __restrict__ bias,
                                    float* __restrict__ out, int n)
{
    long gid = (long)blockIdx.x * blockDim.x + threadIdx.x;
    long i = gid >> 7;
    int  c = (int)(gid & 127);
    if (i < n) {
        float dv = dinv[i];
        out[i * HDIM + c] = h[i * HDIM + c] * dv * dv + bias[c];
    }
}

// ---------------------------------------------------------------------------
// Edge scatter: wave per edge, lane handles 4 channels (float4 + 4 atomics).
// Working set (~77 MB/step) is L2-resident on MI455X's 192 MB L2.
// ---------------------------------------------------------------------------
__global__ void edge_scatter_kernel(const float* __restrict__ h,
                                    const float* __restrict__ dinv,
                                    const int* __restrict__ src,
                                    const int* __restrict__ dst,
                                    float* __restrict__ out, int ne)
{
    long e    = (long)blockIdx.x * 8 + (threadIdx.x >> 5);
    int  lane = threadIdx.x & 31;
    if (e >= ne) return;
    int s = src[e], d = dst[e];
    float nrm = dinv[s] * dinv[d];
    const float4 v = *(const float4*)(h + (long)s * HDIM + lane * 4);
    float* o = out + (long)d * HDIM + lane * 4;
    atomicAdd(o + 0, v.x * nrm);
    atomicAdd(o + 1, v.y * nrm);
    atomicAdd(o + 2, v.z * nrm);
    atomicAdd(o + 3, v.w * nrm);
}

// ---------------------------------------------------------------------------
// Pool: g[c] += sum_i relu(h[i][c])   (mean's 1/N applied in GRU kernel)
// ---------------------------------------------------------------------------
__global__ void pool_relu_kernel(const float* __restrict__ h,
                                 float* __restrict__ g, int n)
{
    __shared__ float red[256];
    int c    = threadIdx.x & 127;
    int half = threadIdx.x >> 7;
    float s = 0.f;
    for (long r = (long)blockIdx.x * 2 + half; r < n; r += (long)gridDim.x * 2)
        s += fmaxf(h[r * HDIM + c], 0.f);
    red[threadIdx.x] = s;
    __syncthreads();
    if (threadIdx.x < 128)
        atomicAdd(&g[c], red[threadIdx.x] + red[threadIdx.x + 128]);
}

__global__ void zero_kernel(float* __restrict__ p, int n) {
    int i = blockIdx.x * blockDim.x + threadIdx.x;
    if (i < n) p[i] = 0.f;
}

// ---------------------------------------------------------------------------
// GRU scan (T=12, H=128) + head. Single block, 128 threads.
// ---------------------------------------------------------------------------
__launch_bounds__(128)
__global__ void gru_head_kernel(const float* __restrict__ gseq,   // T*128 sums
                                const float* __restrict__ Wih,    // (3H,H)
                                const float* __restrict__ Whh,    // (3H,H)
                                const float* __restrict__ bih,
                                const float* __restrict__ bhh,
                                const float* __restrict__ Whead,  // (C,H)
                                const float* __restrict__ bhead,
                                float* __restrict__ out, float inv_n)
{
    __shared__ float xs[128], hs[128];
    int tid = threadIdx.x;
    hs[tid] = 0.f;
    __syncthreads();
    for (int t = 0; t < TSTEPS; ++t) {
        xs[tid] = gseq[t * 128 + tid] * inv_n;
        __syncthreads();
        float ir = bih[tid], iz = bih[128 + tid], in_ = bih[256 + tid];
        float hr = bhh[tid], hz = bhh[128 + tid], hn  = bhh[256 + tid];
        for (int k = 0; k < 128; ++k) {
            float x = xs[k], hh = hs[k];
            ir  += Wih[tid * 128 + k] * x;
            iz  += Wih[(128 + tid) * 128 + k] * x;
            in_ += Wih[(256 + tid) * 128 + k] * x;
            hr  += Whh[tid * 128 + k] * hh;
            hz  += Whh[(128 + tid) * 128 + k] * hh;
            hn  += Whh[(256 + tid) * 128 + k] * hh;
        }
        float r = 1.f / (1.f + expf(-(ir + hr)));
        float z = 1.f / (1.f + expf(-(iz + hz)));
        float nn = tanhf(in_ + r * hn);
        float hnew = (1.f - z) * nn + z * hs[tid];
        __syncthreads();
        hs[tid] = hnew;
        __syncthreads();
    }
    float acc = bhead[tid];
    for (int k = 0; k < 128; ++k) acc += Whead[tid * 128 + k] * hs[k];
    out[tid] = acc;
}

// ---------------------------------------------------------------------------
extern "C" void kernel_launch(void* const* d_in, const int* in_sizes, int n_in,
                              void* d_out, int out_size, void* d_ws, size_t ws_size,
                              hipStream_t stream)
{
    const float* x_seq  = (const float*)d_in[0];   // (T,N,128)
    const int*   ei_seq = (const int*)  d_in[1];   // (T,2,E)
    const float* W1     = (const float*)d_in[2];
    const float* b1     = (const float*)d_in[3];
    const float* W2     = (const float*)d_in[4];
    const float* b2     = (const float*)d_in[5];
    const float* Wih    = (const float*)d_in[6];
    const float* Whh    = (const float*)d_in[7];
    const float* bih    = (const float*)d_in[8];
    const float* bhh    = (const float*)d_in[9];
    const float* Whead  = (const float*)d_in[10];
    const float* bhead  = (const float*)d_in[11];
    float* out = (float*)d_out;

    const int N = in_sizes[0] / (TSTEPS * HDIM);
    const int E = in_sizes[1] / (TSTEPS * 2);

    // workspace layout (floats)
    float* deg  = (float*)d_ws;
    float* dinv = deg + N;
    float* gseq = dinv + N;                          // TSTEPS*128
    size_t off  = (size_t)(2 * N + TSTEPS * 128 + 64) & ~(size_t)15;
    float* bufA = (float*)d_ws + off;                // N*128
    float* bufB = bufA + (size_t)N * HDIM;           // N*128

    const int thr = 256;
    const int gemm_blocks = (N + 127) / 128;
    const int nc_blocks   = (int)(((long)N * HDIM + thr - 1) / thr);
    const int n_blocks    = (N + thr - 1) / thr;
    const int e_blocks    = (E + thr - 1) / thr;
    const int sc_blocks   = (int)(((long)E * 32 + thr - 1) / thr);

    zero_kernel<<<(TSTEPS * 128 + 255) / 256, 256, 0, stream>>>(gseq, TSTEPS * 128);

    for (int t = 0; t < TSTEPS; ++t) {
        const float* Xt  = x_seq + (size_t)t * N * HDIM;
        const int*   src = ei_seq + (size_t)t * 2 * E;
        const int*   dst = src + E;

        // symmetric normalization
        init_deg_kernel<<<n_blocks, thr, 0, stream>>>(deg, N);
        deg_edges_kernel<<<e_blocks, thr, 0, stream>>>(deg, dst, E);
        dinv_kernel<<<n_blocks, thr, 0, stream>>>(deg, dinv, N);

        // layer 1: h1 = X @ W1 ; out1 = scatter(norm*h1) + b1
        gcn_gemm_wmma<<<gemm_blocks, thr, 0, stream>>>(Xt, W1, bufA, N, 0);
        scatter_init_kernel<<<nc_blocks, thr, 0, stream>>>(bufA, dinv, b1, bufB, N);
        edge_scatter_kernel<<<sc_blocks, thr, 0, stream>>>(bufA, dinv, src, dst, bufB, E);

        // layer 2: h2 = relu(out1) @ W2 ; out2 = scatter(norm*h2) + b2
        gcn_gemm_wmma<<<gemm_blocks, thr, 0, stream>>>(bufB, W2, bufA, N, 1);
        scatter_init_kernel<<<nc_blocks, thr, 0, stream>>>(bufA, dinv, b2, bufB, N);
        edge_scatter_kernel<<<sc_blocks, thr, 0, stream>>>(bufA, dinv, src, dst, bufB, E);

        // g[t] = mean_i relu(out2[i])  (sum here, 1/N in GRU kernel)
        pool_relu_kernel<<<512, thr, 0, stream>>>(bufB, gseq + t * 128, N);
    }

    gru_head_kernel<<<1, 128, 0, stream>>>(gseq, Wih, Whh, bih, bhh,
                                           Whead, bhead, out, 1.0f / (float)N);
}
